// BigramLanguageModel_18330920419414
// MI455X (gfx1250) — matrix operand order repaired and checked by reading, not compiled, not run on hardware
//
#include <hip/hip_runtime.h>
#include <hip/hip_bf16.h>

typedef _Float16 h16;
typedef __attribute__((ext_vector_type(16))) _Float16 v16h;
typedef __attribute__((ext_vector_type(8)))  _Float16 v8h;
typedef __attribute__((ext_vector_type(8)))  float    v8f;
typedef __attribute__((ext_vector_type(4)))  unsigned int u32x4;
typedef __attribute__((ext_vector_type(8)))  int i32x8;
typedef __attribute__((ext_vector_type(4)))  int i32x4;

#define B_  64
#define T_  128
#define C_  384
#define H_  6
#define DH_ 64
#define FF_ 1536
#define V_  25000
#define L_  6
#define NT_ (B_*T_)

#define KC_  64   // K-chunk staged per TDM transfer (all K are multiples of 64)
#define KCP_ 72   // LDS pitch in halfs: 64 + 8 pad (TDM pad: 4 DWORDs per 32 DWORDs)

// ---------------------------------------------------------------------------
// A-fragment loader, CDNA5 16-bit WMMA A layout (16x32, ISA 7.12.2):
// lanes 0-15: row M=lane, K base 0; lanes 16-31: same rows, K base 8.
// v16h elems 0..7 = A[row, kb..kb+7], elems 8..15 = A[row, kb+16..kb+23].
// ---------------------------------------------------------------------------
static __device__ __forceinline__
v16h load_a_frag(const h16* __restrict__ A, int lda, int m0, int k0, int lane) {
    int row = m0 + (lane & 15);
    int kb  = k0 + ((lane >> 4) << 3);
    const h16* p = A + (size_t)row * lda + kb;
    v8h lo = *reinterpret_cast<const v8h*>(p);
    v8h hi = *reinterpret_cast<const v8h*>(p + 16);
    v16h r;
#pragma unroll
    for (int i = 0; i < 8; ++i) { r[i] = lo[i]; r[8 + i] = hi[i]; }
    return r;
}

// B-fragment from the LDS-staged tile (N-major, pitch KCP_):
// lanes 0-15: col n, K base ks; lanes 16-31: K base ks+8; elems 0..7 = K kb..kb+7, 8..15 = K kb+16..kb+23.
static __device__ __forceinline__
v16h load_b_lds(const h16* tile, int n0loc, int ks, int lane) {
    int col = n0loc + (lane & 15);
    int kb  = ks + ((lane >> 4) << 3);
    const h16* p = tile + col * KCP_ + kb;
    v8h lo = *reinterpret_cast<const v8h*>(p);
    v8h hi = *reinterpret_cast<const v8h*>(p + 16);
    v16h r;
#pragma unroll
    for (int i = 0; i < 8; ++i) { r[i] = lo[i]; r[8 + i] = hi[i]; }
    return r;
}

// ---------------------------------------------------------------------------
// TDM: DMA one B tile chunk [64 cols x KC_ halfs] (N-major, row stride ldb)
// from global into LDS with 16B padding every 128B (pitch -> KCP_ halfs).
// tensor_dim1 = ncols clamps the V tail: OOB rows are zero-filled by the TDM.
// Descriptor packing per CDNA5 ISA ch.8 (D# groups 0/1; 2D tile -> groups 2/3 zero).
// ---------------------------------------------------------------------------
static __device__ __forceinline__
void tdm_load_b_tile(const h16* src, int ldb, int ncols, unsigned lds_addr) {
    unsigned long long ga = (unsigned long long)(const void*)src;
    u32x4 g0;
    g0[0] = 1u;                                              // count=1, user descriptor
    g0[1] = lds_addr;                                        // lds_addr [63:32]
    g0[2] = (unsigned)(ga & 0xffffffffu);                    // global_addr lo
    g0[3] = (unsigned)((ga >> 32) & 0x1ffffffu) | (2u << 30);// global_addr hi | type=2
    i32x8 g1;
    g1[0] = (int)((1u << 16)        // data_size = 2 bytes
                | (1u << 20)        // pad_enable
                | (4u << 22)        // pad_interval: every 32 DWORDs (128B)
                | (3u << 25));      // pad_amount: 4 DWORDs (16B)
    g1[1] = (int)(((unsigned)KC_ & 0xffffu) << 16);          // tensor_dim0 lo16 @[63:48]
    g1[2] = (int)(((unsigned)ncols & 0xffffu) << 16);        // tdim0 hi=0, tensor_dim1 lo16
    g1[3] = (int)((((unsigned)ncols >> 16) & 0xffffu)        // tensor_dim1 hi16
                | ((unsigned)KC_ << 16));                    // tile_dim0 @[127:112]
    g1[4] = 64;                                              // tile_dim1=64, tile_dim2=0
    g1[5] = (int)(unsigned)ldb;                              // tensor_dim0_stride lo32
    g1[6] = 0;                                               // stride hi, dim1_stride lo
    g1[7] = 0;
    i32x4 z4 = {0, 0, 0, 0};
#if __clang_major__ >= 23
    i32x8 z8 = {0, 0, 0, 0, 0, 0, 0, 0};
    __builtin_amdgcn_tensor_load_to_lds(g0, g1, z4, z4, z8, 0);
#else
    __builtin_amdgcn_tensor_load_to_lds(g0, g1, z4, z4, 0);
#endif
}

// ---------------------------------------------------------------------------
// Batched WMMA GEMM: C[M,N] = A[M,K] * B[K,N] (B given N-major as Bt).
// 256 threads = 8 waves; wave tile 32x32 (2x2 frags); WG tile 128x64.
// B tile TDM-staged into LDS, double-buffered; A streamed from global.
// blockIdx.z -> (z/Hdiv, z%Hdiv) with independent strides for (batch, head).
// ---------------------------------------------------------------------------
template<bool BIAS, bool RELU, bool RESID, bool OUT32, bool OUT16>
__global__ void __launch_bounds__(256)
gemm_wmma(const h16* __restrict__ A, const h16* __restrict__ Bt,
          const float* __restrict__ bias,
          float* __restrict__ C32, h16* __restrict__ C16,
          int lda, int ldb, int ldc, int ldc16,
          int N, int K, int Hdiv,
          long long sAb, long long sAh,
          long long sBb, long long sBh,
          long long sCb, long long sCh,
          long long sDb, long long sDh)
{
    __shared__ h16 ldsB[2][64 * KCP_];

    const int lane = threadIdx.x & 31;
    const int wave = threadIdx.x >> 5;
    const int m0   = blockIdx.y * 128 + (wave >> 1) * 32;
    const int nblk = blockIdx.x * 64;
    const int n0loc = (wave & 1) * 32;

    const int z  = blockIdx.z;
    const int zb = z / Hdiv, zh = z % Hdiv;
    A  += zb * sAb + zh * sAh;
    Bt += zb * sBb + zh * sBh;
    if (OUT32) C32 += zb * sCb + zh * sCh;
    if (OUT16) C16 += zb * sDb + zh * sDh;

    const h16* btile = Bt + (size_t)nblk * ldb;
    const int  ncols = N - nblk;
    const unsigned ldsA0 = (unsigned)(size_t)(&ldsB[0][0]);
    const unsigned ldsA1 = (unsigned)(size_t)(&ldsB[1][0]);

    const int nchunks = K / KC_;
    if (wave == 0)
        tdm_load_b_tile(btile, ldb, ncols, ldsA0);

    v8f acc[2][2] = {};

    for (int c = 0; c < nchunks; ++c) {
        if (wave == 0) {
            if (c + 1 < nchunks) {
                tdm_load_b_tile(btile + (c + 1) * KC_, ldb, ncols, (c & 1) ? ldsA0 : ldsA1);
                __builtin_amdgcn_s_wait_tensorcnt(1);   // chunk c arrived
            } else {
                __builtin_amdgcn_s_wait_tensorcnt(0);
            }
        }
        __syncthreads();                                // chunk c visible to all waves
        const h16* tile = &ldsB[c & 1][0];
        const int kbase = c * KC_;
#pragma unroll
        for (int ks = 0; ks < KC_; ks += 32) {
            v16h a0 = load_a_frag(A, lda, m0,      kbase + ks, lane);
            v16h a1 = load_a_frag(A, lda, m0 + 16, kbase + ks, lane);
            v16h b0 = load_b_lds(tile, n0loc,      ks, lane);
            v16h b1 = load_b_lds(tile, n0loc + 16, ks, lane);
            acc[0][0] = __builtin_amdgcn_wmma_f32_16x16x32_f16(false, a0, false, b0, (short)0, acc[0][0], false, false);
            acc[0][1] = __builtin_amdgcn_wmma_f32_16x16x32_f16(false, a0, false, b1, (short)0, acc[0][1], false, false);
            acc[1][0] = __builtin_amdgcn_wmma_f32_16x16x32_f16(false, a1, false, b0, (short)0, acc[1][0], false, false);
            acc[1][1] = __builtin_amdgcn_wmma_f32_16x16x32_f16(false, a1, false, b1, (short)0, acc[1][1], false, false);
        }
        __syncthreads();                                // buffer c&1 free for reissue
    }

    // D layout: lanes 0-15: N=lane, M=i per VGPR i; lanes 16-31: N=lane-16, M=8+i.
#pragma unroll
    for (int mi = 0; mi < 2; ++mi) {
#pragma unroll
        for (int ni = 0; ni < 2; ++ni) {
            int col = nblk + n0loc + ni * 16 + (lane & 15);
            if (col >= N) continue;
            int rowb = m0 + mi * 16 + ((lane >> 4) << 3);
            float bv = BIAS ? bias[col] : 0.0f;
#pragma unroll
            for (int i = 0; i < 8; ++i) {
                int row = rowb + i;
                float v = acc[mi][ni][i] + bv;
                if (RELU)  v = v > 0.0f ? v : 0.0f;
                if (RESID) v += C32[(size_t)row * ldc + col];
                if (OUT32) C32[(size_t)row * ldc + col] = v;
                if (OUT16) C16[(size_t)row * ldc16 + col] = (h16)v;
            }
        }
    }
}

// ---------------------------------------------------------------------------
__global__ void __launch_bounds__(256)
embed_kernel(const int* __restrict__ idx, const float* __restrict__ tok,
             const float* __restrict__ pos, float* __restrict__ x, int total)
{
    int i = blockIdx.x * 256 + threadIdx.x;
    if (i >= total) return;
    int c  = i % C_;
    int bt = i / C_;
    int t  = bt % T_;
    x[i] = tok[(size_t)idx[bt] * C_ + c] + pos[(size_t)t * C_ + c];
}

__global__ void __launch_bounds__(256)
convert_transpose(const float* __restrict__ in, h16* __restrict__ out, int K, int N)
{
    int i = blockIdx.x * 256 + threadIdx.x;
    if (i >= K * N) return;
    int n = i / K, k = i % K;
    out[(size_t)n * K + k] = (h16)in[(size_t)k * N + n];
}

// v[B,T,H,DH] f16 -> vT[B,H,DH,T] f16
__global__ void __launch_bounds__(256)
transpose_v_kernel(const h16* __restrict__ v, h16* __restrict__ vT)
{
    int i = blockIdx.x * 256 + threadIdx.x;
    if (i >= NT_ * C_) return;
    int d = i & (DH_ - 1);
    int r = i >> 6;
    int h = r % H_;
    int r2 = r / H_;
    int t = r2 & (T_ - 1);
    int b = r2 >> 7;
    vT[(((size_t)b * H_ + h) * DH_ + d) * T_ + t] = v[i];
}

__global__ void __launch_bounds__(128)
layernorm_kernel(const float* __restrict__ x, const float* __restrict__ g,
                 const float* __restrict__ b, h16* __restrict__ out, int nrows)
{
    int row  = blockIdx.x * 4 + (threadIdx.x >> 5);
    int lane = threadIdx.x & 31;
    if (row >= nrows) return;
    const float* xr = x + (size_t)row * C_;
    float vals[12];
    float s = 0.0f;
#pragma unroll
    for (int j = 0; j < 12; ++j) { vals[j] = xr[lane + 32 * j]; s += vals[j]; }
#pragma unroll
    for (int o = 16; o > 0; o >>= 1) s += __shfl_xor(s, o, 32);
    float mu = s * (1.0f / C_);
    float vs = 0.0f;
#pragma unroll
    for (int j = 0; j < 12; ++j) { float d = vals[j] - mu; vs += d * d; }
#pragma unroll
    for (int o = 16; o > 0; o >>= 1) vs += __shfl_xor(vs, o, 32);
    float rstd = rsqrtf(vs * (1.0f / C_) + 1e-5f);
    h16* outr = out + (size_t)row * C_;
#pragma unroll
    for (int j = 0; j < 12; ++j) {
        int c = lane + 32 * j;
        outr[c] = (h16)((vals[j] - mu) * rstd * g[c] + b[c]);
    }
}

__global__ void __launch_bounds__(128)
softmax_causal(const float* __restrict__ scores, h16* __restrict__ att, float scale)
{
    int row  = blockIdx.x * 4 + (threadIdx.x >> 5);
    int lane = threadIdx.x & 31;
    int t = row & (T_ - 1);
    const float* sr = scores + (size_t)row * T_;
    float v[4];
    float mx = -1e30f;
#pragma unroll
    for (int j = 0; j < 4; ++j) {
        int c = lane * 4 + j;
        float val = sr[c] * scale;
        v[j] = (c <= t) ? val : -1e30f;
        mx = fmaxf(mx, v[j]);
    }
#pragma unroll
    for (int o = 16; o > 0; o >>= 1) mx = fmaxf(mx, __shfl_xor(mx, o, 32));
    float s = 0.0f;
#pragma unroll
    for (int j = 0; j < 4; ++j) { v[j] = __expf(v[j] - mx); s += v[j]; }
#pragma unroll
    for (int o = 16; o > 0; o >>= 1) s += __shfl_xor(s, o, 32);
    float inv = 1.0f / s;
    h16* ar = att + (size_t)row * T_;
#pragma unroll
    for (int j = 0; j < 4; ++j) ar[lane * 4 + j] = (h16)(v[j] * inv);
}

__global__ void __launch_bounds__(256)
nll_kernel(const float* __restrict__ logits, const int* __restrict__ tgt,
           float* __restrict__ nll)
{
    __shared__ float red[256];
    int row = blockIdx.x;
    const float* lr = logits + (size_t)row * V_;
    float mx = -1e30f;
    for (int c = threadIdx.x; c < V_; c += 256) mx = fmaxf(mx, lr[c]);
    red[threadIdx.x] = mx; __syncthreads();
    for (int s = 128; s > 0; s >>= 1) {
        if (threadIdx.x < s) red[threadIdx.x] = fmaxf(red[threadIdx.x], red[threadIdx.x + s]);
        __syncthreads();
    }
    mx = red[0]; __syncthreads();
    float sum = 0.0f;
    for (int c = threadIdx.x; c < V_; c += 256) sum += __expf(lr[c] - mx);
    red[threadIdx.x] = sum; __syncthreads();
    for (int s = 128; s > 0; s >>= 1) {
        if (threadIdx.x < s) red[threadIdx.x] += red[threadIdx.x + s];
        __syncthreads();
    }
    if (threadIdx.x == 0)
        nll[row] = -(lr[tgt[row]] - mx - __logf(red[0]));
}

__global__ void __launch_bounds__(256)
loss_kernel(const float* __restrict__ nll, float* __restrict__ out)
{
    __shared__ float red[256];
    float s = 0.0f;
    for (int i = threadIdx.x; i < NT_; i += 256) s += nll[i];
    red[threadIdx.x] = s; __syncthreads();
    for (int st = 128; st > 0; st >>= 1) {
        if (threadIdx.x < st) red[threadIdx.x] += red[threadIdx.x + st];
        __syncthreads();
    }
    if (threadIdx.x == 0) *out = red[0] * (1.0f / NT_);
}

// ---------------------------------------------------------------------------
extern "C" void kernel_launch(void* const* d_in, const int* in_sizes, int n_in,
                              void* d_out, int out_size, void* d_ws, size_t ws_size,
                              hipStream_t stream)
{
    (void)in_sizes; (void)n_in; (void)out_size; (void)ws_size;
    const int*   idx     = (const int*)  d_in[0];
    const int*   targets = (const int*)  d_in[1];
    const float* tok_emb = (const float*)d_in[2];
    const float* pos_emb = (const float*)d_in[3];
    const float* wq  = (const float*)d_in[4];
    const float* wk  = (const float*)d_in[5];
    const float* wv  = (const float*)d_in[6];
    const float* wo  = (const float*)d_in[7];
    const float* bo  = (const float*)d_in[8];
    const float* ln1_g = (const float*)d_in[9];
    const float* ln1_b = (const float*)d_in[10];
    const float* ln2_g = (const float*)d_in[11];
    const float* ln2_b = (const float*)d_in[12];
    const float* w1  = (const float*)d_in[13];
    const float* b1  = (const float*)d_in[14];
    const float* w2  = (const float*)d_in[15];
    const float* b2  = (const float*)d_in[16];
    const float* lnf_g = (const float*)d_in[17];
    const float* lnf_b = (const float*)d_in[18];
    const float* lm_w  = (const float*)d_in[19];
    const float* lm_b  = (const float*)d_in[20];

    float* logits = (float*)d_out;

    char* p = (char*)d_ws;
    auto carve = [&](size_t bytes) -> void* {
        void* r = (void*)p;
        p += (bytes + 255) & ~(size_t)255;
        return r;
    };
    float* x      = (float*)carve((size_t)NT_ * C_ * 4);
    h16*   hbuf   = (h16*)  carve((size_t)NT_ * C_ * 2);
    h16*   q16    = (h16*)  carve((size_t)NT_ * C_ * 2);
    h16*   k16    = (h16*)  carve((size_t)NT_ * C_ * 2);
    h16*   v16b   = (h16*)  carve((size_t)NT_ * C_ * 2);
    h16*   vT16   = (h16*)  carve((size_t)NT_ * C_ * 2);
    h16*   o16    = (h16*)  carve((size_t)NT_ * C_ * 2);
    float* scores = (float*)carve((size_t)B_ * H_ * T_ * T_ * 4);
    h16*   att    = (h16*)  carve((size_t)B_ * H_ * T_ * T_ * 2);
    h16*   ff16   = (h16*)  carve((size_t)NT_ * FF_ * 2);
    h16*   wqT    = (h16*)  carve((size_t)C_ * C_ * 2);
    h16*   wkT    = (h16*)  carve((size_t)C_ * C_ * 2);
    h16*   wvT    = (h16*)  carve((size_t)C_ * C_ * 2);
    h16*   woT    = (h16*)  carve((size_t)C_ * C_ * 2);
    h16*   w1T    = (h16*)  carve((size_t)C_ * FF_ * 2);
    h16*   w2T    = (h16*)  carve((size_t)C_ * FF_ * 2);
    h16*   lmwT   = (h16*)  carve((size_t)V_ * C_ * 2);
    float* nllbuf = (float*)carve((size_t)NT_ * 4);

    const dim3 blk256(256);
    const dim3 blk128(128);
    const int cvtCC = (C_ * C_ + 255) / 256;
    const int cvtCF = (C_ * FF_ + 255) / 256;

    embed_kernel<<<(NT_ * C_ + 255) / 256, blk256, 0, stream>>>(idx, tok_emb, pos_emb, x, NT_ * C_);

    const size_t CC = (size_t)C_ * C_;
    const size_t CF = (size_t)C_ * FF_;

    for (int l = 0; l < L_; ++l) {
        convert_transpose<<<cvtCC, blk256, 0, stream>>>(wq + l * CC, wqT, C_, C_);
        convert_transpose<<<cvtCC, blk256, 0, stream>>>(wk + l * CC, wkT, C_, C_);
        convert_transpose<<<cvtCC, blk256, 0, stream>>>(wv + l * CC, wvT, C_, C_);
        convert_transpose<<<cvtCC, blk256, 0, stream>>>(wo + l * CC, woT, C_, C_);
        convert_transpose<<<cvtCF, blk256, 0, stream>>>(w1 + l * CF, w1T, C_, FF_);
        convert_transpose<<<cvtCF, blk256, 0, stream>>>(w2 + l * CF, w2T, FF_, C_);

        // h = LN1(x)  (f16)
        layernorm_kernel<<<NT_ / 4, blk128, 0, stream>>>(x, ln1_g + l * C_, ln1_b + l * C_, hbuf, NT_);

        // q,k,v = h @ W  -> f16 [B,T,H,DH]
        dim3 gqkv(C_ / 64, NT_ / 128, 1);
        gemm_wmma<false,false,false,false,true><<<gqkv, blk256, 0, stream>>>(
            hbuf, wqT, nullptr, nullptr, q16, C_, C_, 0, C_, C_, C_, 1, 0,0,0,0,0,0,0,0);
        gemm_wmma<false,false,false,false,true><<<gqkv, blk256, 0, stream>>>(
            hbuf, wkT, nullptr, nullptr, k16, C_, C_, 0, C_, C_, C_, 1, 0,0,0,0,0,0,0,0);
        gemm_wmma<false,false,false,false,true><<<gqkv, blk256, 0, stream>>>(
            hbuf, wvT, nullptr, nullptr, v16b, C_, C_, 0, C_, C_, C_, 1, 0,0,0,0,0,0,0,0);

        transpose_v_kernel<<<(NT_ * C_ + 255) / 256, blk256, 0, stream>>>(v16b, vT16);

        // scores[b,h] = q[b,:,h,:] @ k[b,:,h,:]^T   (k row-major == N-major operand)
        dim3 gsc(T_ / 64, T_ / 128, B_ * H_);
        gemm_wmma<false,false,false,true,false><<<gsc, blk256, 0, stream>>>(
            q16, k16, nullptr, scores, nullptr, C_, C_, T_, 0, T_, DH_, H_,
            (long long)T_ * C_, (long long)DH_,
            (long long)T_ * C_, (long long)DH_,
            (long long)H_ * T_ * T_, (long long)T_ * T_, 0, 0);

        softmax_causal<<<(B_ * H_ * T_) / 4, blk128, 0, stream>>>(scores, att, 0.125f);

        // o[b,:,h,:] = att[b,h] @ v[b,:,h,:]   (vT is N-major operand)
        dim3 gav(1, 1, B_ * H_);
        gemm_wmma<false,false,false,false,true><<<gav, blk256, 0, stream>>>(
            att, vT16, nullptr, nullptr, o16, T_, T_, 0, C_, DH_, T_, H_,
            (long long)H_ * T_ * T_, (long long)T_ * T_,
            (long long)H_ * DH_ * T_, (long long)DH_ * T_,
            0, 0, (long long)T_ * C_, (long long)DH_);

        // x = x + o @ wo + bo   (residual in-place)
        dim3 gwo(C_ / 64, NT_ / 128, 1);
        gemm_wmma<true,false,true,true,false><<<gwo, blk256, 0, stream>>>(
            o16, woT, bo + l * C_, x, nullptr, C_, C_, C_, 0, C_, C_, 1, 0,0,0,0,0,0,0,0);

        // h2 = LN2(x)
        layernorm_kernel<<<NT_ / 4, blk128, 0, stream>>>(x, ln2_g + l * C_, ln2_b + l * C_, hbuf, NT_);

        // ff = relu(h2 @ w1 + b1)  (f16)
        dim3 gf1(FF_ / 64, NT_ / 128, 1);
        gemm_wmma<true,true,false,false,true><<<gf1, blk256, 0, stream>>>(
            hbuf, w1T, b1 + l * FF_, nullptr, ff16, C_, C_, 0, FF_, FF_, C_, 1, 0,0,0,0,0,0,0,0);

        // x = x + ff @ w2 + b2
        dim3 gf2(C_ / 64, NT_ / 128, 1);
        gemm_wmma<true,false,true,true,false><<<gf2, blk256, 0, stream>>>(
            ff16, w2T, b2 + l * C_, x, nullptr, FF_, FF_, C_, 0, C_, FF_, 1, 0,0,0,0,0,0,0,0);
    }

    // final LN -> f16, LM head -> logits f32 in d_out
    layernorm_kernel<<<NT_ / 4, blk128, 0, stream>>>(x, lnf_g, lnf_b, hbuf, NT_);
    convert_transpose<<<(int)(((size_t)C_ * V_ + 255) / 256), blk256, 0, stream>>>(lm_w, lmwT, C_, V_);
    dim3 glm((V_ + 63) / 64, NT_ / 128, 1);
    gemm_wmma<true,false,false,true,false><<<glm, blk256, 0, stream>>>(
        hbuf, lmwT, lm_b, logits, nullptr, C_, C_, V_, 0, V_, C_, 1, 0,0,0,0,0,0,0,0);

    // loss
    nll_kernel<<<NT_, blk256, 0, stream>>>(logits, targets, nllbuf);
    loss_kernel<<<1, blk256, 0, stream>>>(nllbuf, logits + (size_t)NT_ * V_);
}
